// FP8Linear_75857712382560
// MI455X (gfx1250) — 8-bit matrix operand order repaired and checked by reading, compiled, not run on hardware
//
#include <hip/hip_runtime.h>
#include <hip/hip_bf16.h>


typedef int   v16i __attribute__((ext_vector_type(16)));
typedef float v8f  __attribute__((ext_vector_type(8)));

#define TILE_BYTES 2048
#define FP8_MAX_F 448.0f
#define EPS_F 1e-12f

#define LDS_BUF_BYTES 49152
#define LDS_A_BYTES   32768
#define EPI_STRIDE    144

__device__ __forceinline__ float bf16lo(unsigned int u) {
    return __uint_as_float(u << 16);
}
__device__ __forceinline__ float bf16hi(unsigned int u) {
    return __uint_as_float(u & 0xFFFF0000u);
}
__device__ __forceinline__ unsigned short f32_to_bf16(float f) {
    unsigned int u = __float_as_uint(f);
    u += 0x7FFFu + ((u >> 16) & 1u);
    return (unsigned short)(u >> 16);
}

__device__ __forceinline__ unsigned int f32_to_e4m3(float f) {
    float a = fminf(fmaxf(f, -FP8_MAX_F), FP8_MAX_F);
    unsigned int u = __float_as_uint(a);
    unsigned int sign = (u >> 24) & 0x80u;
    float mag = fabsf(a);
    unsigned int r;
    if (mag < 0.015625f) {
        r = (unsigned int)__builtin_rintf(mag * 512.0f);
    } else {
        unsigned int m = u & 0x7FFFFFFFu;
        m += 0x0007FFFFu + ((m >> 20) & 1u);
        unsigned int exp  = m >> 23;
        unsigned int mant = (m >> 20) & 7u;
        int e8 = (int)exp - 120;
        r = ((unsigned int)e8 << 3) | mant;
        if (r > 0x7Eu) r = 0x7Eu;
    }
    return sign | r;
}

__device__ __forceinline__ size_t a_swizzle(int m, int k, int MT) {
    int mt = m >> 4;
    int kt = k >> 7;
    int Kt = k & 127;
    int h  = Kt >> 6;
    int K2 = Kt & 63;
    int v  = (((K2 >> 5) << 2) | ((K2 & 15) >> 2)) + (h << 3);
    int g  = (K2 >> 4) & 1;
    int lane = (m & 15) | (g << 4);
    return (size_t)(kt * MT + mt) * TILE_BYTES + lane * 64 + v * 4;
}
__device__ __forceinline__ size_t b_swizzle(int n, int k, int NT) {
    int nt = n >> 4;
    int kt = k >> 7;
    int Kt = k & 127;
    int v  = ((Kt >> 5) << 2) | ((Kt & 15) >> 2);
    int g  = (Kt >> 4) & 1;
    int lane = (n & 15) | (g << 4);
    return (size_t)(kt * NT + nt) * TILE_BYTES + lane * 64 + v * 4;
}

template <int OFF>
__device__ __forceinline__ void async_cp16(unsigned lds_off, const void* g) {
    asm volatile("global_load_async_to_lds_b128 %0, %1, off offset:%c2"
                 :: "v"(lds_off), "v"((unsigned long long)(size_t)g), "i"(OFF)
                 : "memory");
}
__device__ __forceinline__ void wait_async0() {
    asm volatile("s_wait_asynccnt 0x0" ::: "memory");
}

__global__ void k_init(unsigned int* wAbs) {
    if (threadIdx.x == 0) *wAbs = 0u;
}

__global__ __launch_bounds__(256)
void k_wabsmax(const unsigned int* __restrict__ w, unsigned int* __restrict__ wAbs,
               long long n_uint4) {
    __shared__ float red[256];
    int tid = threadIdx.x;
    float mx = 0.0f;
    const uint4* p = (const uint4*)w;
    for (long long i = (long long)blockIdx.x * 256 + tid; i < n_uint4;
         i += (long long)gridDim.x * 256) {
        uint4 q = p[i];
        unsigned int U[4] = {q.x, q.y, q.z, q.w};
#pragma unroll
        for (int j = 0; j < 4; ++j) {
            mx = fmaxf(mx, fabsf(bf16lo(U[j])));
            mx = fmaxf(mx, fabsf(bf16hi(U[j])));
        }
    }
    red[tid] = mx;
    __syncthreads();
    for (int s = 128; s > 0; s >>= 1) {
        if (tid < s) red[tid] = fmaxf(red[tid], red[tid + s]);
        __syncthreads();
    }
    if (tid == 0) atomicMax(wAbs, __float_as_uint(red[0]));
}

__global__ __launch_bounds__(256)
void k_wquant(const unsigned short* __restrict__ w,
              const unsigned int* __restrict__ wAbs,
              unsigned char* __restrict__ Bq, int K, int NT) {
    int n   = blockIdx.x;
    int tid = threadIdx.x;
    float scale = fmaxf(__uint_as_float(*wAbs), EPS_F) * (1.0f / FP8_MAX_F);
    float inv   = 1.0f / scale;
    const uint4* row = (const uint4*)(w + (size_t)n * K);
    for (int k0 = tid * 16; k0 < K; k0 += 256 * 16) {
        uint4 u0 = row[(k0 >> 3) + 0];
        uint4 u1 = row[(k0 >> 3) + 1];
        unsigned int U[8] = {u0.x, u0.y, u0.z, u0.w, u1.x, u1.y, u1.z, u1.w};
#pragma unroll
        for (int j = 0; j < 4; ++j) {
            unsigned int a = U[2 * j], b = U[2 * j + 1];
            unsigned int d =  f32_to_e4m3(bf16lo(a) * inv)
                           | (f32_to_e4m3(bf16hi(a) * inv) << 8)
                           | (f32_to_e4m3(bf16lo(b) * inv) << 16)
                           | (f32_to_e4m3(bf16hi(b) * inv) << 24);
            *(unsigned int*)(Bq + b_swizzle(n, k0 + 4 * j, NT)) = d;
        }
    }
}

__global__ __launch_bounds__(256)
void k_xquant(const unsigned short* __restrict__ x,
              unsigned char* __restrict__ Aq, float* __restrict__ xScale,
              int K, int MT) {
    __shared__ float red[256];
    int m   = blockIdx.x;
    int tid = threadIdx.x;
    const uint4* row = (const uint4*)(x + (size_t)m * K);
    float mx = 0.0f;
    for (int k0 = tid * 16; k0 < K; k0 += 256 * 16) {
        uint4 u0 = row[(k0 >> 3) + 0];
        uint4 u1 = row[(k0 >> 3) + 1];
        unsigned int U[8] = {u0.x, u0.y, u0.z, u0.w, u1.x, u1.y, u1.z, u1.w};
#pragma unroll
        for (int j = 0; j < 8; ++j)
            mx = fmaxf(mx, fmaxf(fabsf(bf16lo(U[j])), fabsf(bf16hi(U[j]))));
    }
    red[tid] = mx;
    __syncthreads();
    for (int s = 128; s > 0; s >>= 1) {
        if (tid < s) red[tid] = fmaxf(red[tid], red[tid + s]);
        __syncthreads();
    }
    float scale = fmaxf(red[0], EPS_F) * (1.0f / FP8_MAX_F);
    if (tid == 0) xScale[m] = scale;
    float inv = 1.0f / scale;
    for (int k0 = tid * 16; k0 < K; k0 += 256 * 16) {
        uint4 u0 = row[(k0 >> 3) + 0];
        uint4 u1 = row[(k0 >> 3) + 1];
        unsigned int U[8] = {u0.x, u0.y, u0.z, u0.w, u1.x, u1.y, u1.z, u1.w};
#pragma unroll
        for (int j = 0; j < 4; ++j) {
            unsigned int a = U[2 * j], b = U[2 * j + 1];
            unsigned int d =  f32_to_e4m3(bf16lo(a) * inv)
                           | (f32_to_e4m3(bf16hi(a) * inv) << 8)
                           | (f32_to_e4m3(bf16lo(b) * inv) << 16)
                           | (f32_to_e4m3(bf16hi(b) * inv) << 24);
            *(unsigned int*)(Aq + a_swizzle(m, k0 + 4 * j, MT)) = d;
        }
    }
}

__global__ __launch_bounds__(256)
void k_gemm(const unsigned char* __restrict__ Aq,
            const unsigned char* __restrict__ Bq,
            const float* __restrict__ xScale,
            const unsigned int* __restrict__ wAbs,
            const unsigned short* __restrict__ bias,
            unsigned short* __restrict__ out,
            int N, int MT, int NT, int KT) {
    __shared__ __align__(128) unsigned char smem[2 * LDS_BUF_BYTES];

    const int tid  = threadIdx.x;
    const int lane = tid & 31;
    const int wave = tid >> 5;
    const int wy   = wave >> 1;
    const int wx   = wave & 1;
    const int mtb  = blockIdx.y * 16;
    const int ntb  = blockIdx.x * 8;

    const unsigned smemBase = (unsigned)(size_t)(void*)&smem[0];

    v8f acc[4][4] = {};

    const unsigned char* aCol = Aq + (size_t)mtb * TILE_BYTES + tid * 128;
    const unsigned char* bCol = Bq + (size_t)ntb * TILE_BYTES + tid * 64;
    const size_t aStride = (size_t)MT * TILE_BYTES;
    const size_t bStride = (size_t)NT * TILE_BYTES;

#define STAGE(kt_, par_)                                                      \
    {                                                                         \
        const unsigned char* as_ = aCol + (size_t)(kt_) * aStride;            \
        const unsigned char* bs_ = bCol + (size_t)(kt_) * bStride;            \
        unsigned ao_ = smemBase + (par_) * LDS_BUF_BYTES + tid * 128;         \
        unsigned bo_ = smemBase + (par_) * LDS_BUF_BYTES + LDS_A_BYTES + tid * 64; \
        async_cp16<0>(ao_, as_);   async_cp16<16>(ao_, as_);                  \
        async_cp16<32>(ao_, as_);  async_cp16<48>(ao_, as_);                  \
        async_cp16<64>(ao_, as_);  async_cp16<80>(ao_, as_);                  \
        async_cp16<96>(ao_, as_);  async_cp16<112>(ao_, as_);                 \
        async_cp16<0>(bo_, bs_);   async_cp16<16>(bo_, bs_);                  \
        async_cp16<32>(bo_, bs_);  async_cp16<48>(bo_, bs_);                  \
    }

    STAGE(0, 0);
    wait_async0();
    __syncthreads();

    for (int kt = 0; kt < KT; ++kt) {
        if (kt + 1 < KT) STAGE(kt + 1, (kt + 1) & 1);

        const unsigned char* bufA = smem + (kt & 1) * LDS_BUF_BYTES;
        const unsigned char* bufB = bufA + LDS_A_BYTES;

        v16i af[4];
#pragma unroll
        for (int ma = 0; ma < 4; ++ma)
            af[ma] = *(const v16i*)(bufA + (wy * 4 + ma) * TILE_BYTES + lane * 64);
#pragma unroll
        for (int nb = 0; nb < 4; ++nb) {
            v16i bf = *(const v16i*)(bufB + (wx * 4 + nb) * TILE_BYTES + lane * 64);
#pragma unroll
            for (int ma = 0; ma < 4; ++ma) {
                acc[ma][nb] = __builtin_amdgcn_wmma_f32_16x16x128_fp8_fp8(
                    af[ma], bf, (short)0, acc[ma][nb], false, false);
            }
        }
        wait_async0();
        __syncthreads();
    }
#undef STAGE

    float wscale = fmaxf(__uint_as_float(*wAbs), EPS_F) * (1.0f / FP8_MAX_F);
    const int g  = lane >> 4;
    const int nl = lane & 15;
    unsigned char* myT = smem + wave * (64 * EPI_STRIDE);

#pragma unroll
    for (int ma = 0; ma < 4; ++ma) {
        float rs[8];
#pragma unroll
        for (int r = 0; r < 8; ++r)
            rs[r] = xScale[(mtb + wy * 4 + ma) * 16 + g * 8 + r] * wscale;
#pragma unroll
        for (int nb = 0; nb < 4; ++nb) {
            int n = (ntb + wx * 4 + nb) * 16 + nl;
            float bv = __uint_as_float(((unsigned int)bias[n]) << 16);
#pragma unroll
            for (int r = 0; r < 8; ++r) {
                float y = acc[ma][nb][r] * rs[r] + bv;
                int row = ma * 16 + g * 8 + r;
                int col = nb * 16 + nl;
                *(unsigned short*)(myT + row * EPI_STRIDE + col * 2) =
                    f32_to_bf16(y);
            }
        }
    }
    const int mBase = (mtb + wy * 4) * 16;
    const size_t nBaseBytes = ((size_t)ntb * 16 + wx * 64) * 2;
    unsigned char* outB = (unsigned char*)out;
#pragma unroll
    for (int p = 0; p < 16; ++p) {
        int row = p * 4 + (lane >> 3);
        int seg = lane & 7;
        int4 v = *(const int4*)(myT + row * EPI_STRIDE + seg * 16);
        *(int4*)(outB + (size_t)(mBase + row) * N * 2 + nBaseBytes + seg * 16) = v;
    }
}

extern "C" void kernel_launch(void* const* d_in, const int* in_sizes, int n_in,
                              void* d_out, int out_size, void* d_ws, size_t ws_size,
                              hipStream_t stream) {
    (void)n_in; (void)out_size; (void)ws_size;
    const unsigned short* x  = (const unsigned short*)d_in[0];
    const unsigned short* w  = (const unsigned short*)d_in[1];
    const unsigned short* bv = (const unsigned short*)d_in[2];

    const int N = in_sizes[2];
    const int K = in_sizes[1] / N;
    const int M = in_sizes[0] / K;
    const int MT = M / 16, NT = N / 16, KT = K / 128;

    unsigned char* ws = (unsigned char*)d_ws;
    unsigned int* wAbs   = (unsigned int*)ws;
    float*        xScale = (float*)(ws + 1024);
    size_t aq_off = (size_t)1 << 20;
    size_t aq_sz  = ((size_t)M * K + 4095) & ~(size_t)4095;
    unsigned char* Aq = ws + aq_off;
    unsigned char* Bq = ws + aq_off + aq_sz;

    k_init<<<1, 64, 0, stream>>>(wAbs);

    long long w_uint4 = (long long)N * K / 8;
    int wb = (int)((w_uint4 + 256 * 8 - 1) / (256 * 8));
    if (wb > 4096) wb = 4096;
    k_wabsmax<<<wb, 256, 0, stream>>>((const unsigned int*)w, wAbs, w_uint4);

    k_wquant<<<N, 256, 0, stream>>>(w, wAbs, Bq, K, NT);
    k_xquant<<<M, 256, 0, stream>>>(x, Aq, xScale, K, MT);

    dim3 grid(N / 128, M / 256);
    k_gemm<<<grid, 256, 0, stream>>>(Aq, Bq, xScale, wAbs, bv,
                                     (unsigned short*)d_out, N, MT, NT, KT);
}
